// GraphSAGE_21114059227292
// MI455X (gfx1250) — matrix operand order repaired and checked by reading, not compiled, not run on hardware
//
#include <hip/hip_runtime.h>
#include <hip/hip_bf16.h>

// ---------------------------------------------------------------------------
// GraphSAGE on MI455X (gfx1250): 3x SAGEConv(128->128)+LeakyReLU, 1x SAGEConv(128->64)
//   agg = segment_mean(x[src] -> dst);  out = agg@Wl + x@Wr + b
// Strategy: f16 activations, f32 atomic segment-sum, v_wmma_f32_16x16x32_f16 GEMMs.
// ---------------------------------------------------------------------------

typedef __attribute__((ext_vector_type(16))) _Float16 v16h;
typedef __attribute__((ext_vector_type(8)))  _Float16 v8h;
typedef __attribute__((ext_vector_type(4)))  _Float16 v4h;
typedef __attribute__((ext_vector_type(8)))  float    v8f;

#define N_NODES 100000
#define N_EDGES 1600000
#define DIM     128
#define DIM_OUT 64

// ---------------- small prep kernels ----------------

// W[K][N] (f32 row-major) -> Wt[N][K] (f16, column of W contiguous)
__global__ void k_transpose_w(const float* __restrict__ W, _Float16* __restrict__ Wt,
                              int K, int Nn) {
    int idx = blockIdx.x * blockDim.x + threadIdx.x;
    if (idx >= K * Nn) return;
    int k = idx / Nn, n = idx % Nn;
    Wt[n * K + k] = (_Float16)W[idx];
}

__global__ void k_to_f16(const float* __restrict__ x, _Float16* __restrict__ h, int n) {
    int idx = blockIdx.x * blockDim.x + threadIdx.x;
    if (idx >= n) return;
    h[idx] = (_Float16)x[idx];
}

// ---------------- aggregation ----------------

__global__ void k_count(const int* __restrict__ dst, int* __restrict__ cnt, int nE) {
    int e = blockIdx.x * blockDim.x + threadIdx.x;
    if (e >= nE) return;
    atomicAdd(&cnt[dst[e]], 1);
}

// one wave32 per edge; each lane handles 4 features -> 4 f32 hw atomics
__global__ void k_scatter(const _Float16* __restrict__ h,
                          const int* __restrict__ src, const int* __restrict__ dst,
                          float* __restrict__ agg, int nE) {
    int t = blockIdx.x * blockDim.x + threadIdx.x;
    int e = t >> 5;
    if (e >= nE) return;
    int lane = t & 31;
    int s = src[e], d = dst[e];
    const _Float16* hp = h + (size_t)s * DIM + lane * 4;
    float* ap = agg + (size_t)d * DIM + lane * 4;
    v4h v = *(const v4h*)hp;
    unsafeAtomicAdd(ap + 0, (float)v[0]);
    unsafeAtomicAdd(ap + 1, (float)v[1]);
    unsafeAtomicAdd(ap + 2, (float)v[2]);
    unsafeAtomicAdd(ap + 3, (float)v[3]);
}

// agg16 = f16( agg32 / max(cnt,1) )
__global__ void k_finalize(const float* __restrict__ agg32, const int* __restrict__ cnt,
                           _Float16* __restrict__ agg16, int n) {
    int idx = blockIdx.x * blockDim.x + threadIdx.x;
    if (idx >= n) return;
    int node = idx >> 7;               // / DIM
    int c = cnt[node];
    float cc = c > 0 ? (float)c : 1.0f;
    agg16[idx] = (_Float16)(agg32[idx] / cc);
}

// ---------------- WMMA GEMM epilogue-fused layer ----------------
// A (16x32 f16) lane layout: group g holds K in [8g,8g+8) and [16+8g, 16+8g+8)
__device__ __forceinline__ v16h ld_a(const _Float16* __restrict__ row, int kb, int g) {
    v8h lo = *(const v8h*)(row + kb + g * 8);
    v8h hi = *(const v8h*)(row + kb + 16 + g * 8);
    return __builtin_shufflevector(lo, hi, 0,1,2,3,4,5,6,7,8,9,10,11,12,13,14,15);
}
// B (32x16 f16) from column-major Wt: lane group g holds K in [8g,8g+8) and [16+8g, 16+8g+8)
__device__ __forceinline__ v16h ld_b(const _Float16* __restrict__ col, int kb, int g) {
    v8h lo = *(const v8h*)(col + kb + g * 8);
    v8h hi = *(const v8h*)(col + kb + 16 + g * 8);
    return __builtin_shufflevector(lo, hi, 0,1,2,3,4,5,6,7,8,9,10,11,12,13,14,15);
}

// One wave per 16x16 output tile.  out = agg16@WlT^T + hIn@WrT^T + b, then
// LeakyReLU(0.1)->f16 (hidden) or raw f32 (final layer).
__global__ void k_sage_gemm(const _Float16* __restrict__ hIn,
                            const _Float16* __restrict__ agg16,
                            const _Float16* __restrict__ WlT,   // [dOut][128] f16
                            const _Float16* __restrict__ WrT,   // [dOut][128] f16
                            const float* __restrict__ bias,     // [dOut]
                            _Float16* __restrict__ hOut,        // hidden path (or null)
                            float* __restrict__ fOut,           // final path (or null)
                            int dOut, int totalTiles) {
    const int K = DIM;
    int wave = threadIdx.x >> 5;
    int lane = threadIdx.x & 31;
    int tile = blockIdx.x * (blockDim.x >> 5) + wave;
    if (tile >= totalTiles) return;            // uniform per wave; EXEC stays all-1s below
    int tilesN = dOut >> 4;
    int tm = tile / tilesN;
    int tn = tile - tm * tilesN;
    int g = lane >> 4, l = lane & 15;

    const _Float16* aggRow = agg16 + (size_t)((tm << 4) + l) * K;
    const _Float16* xRow   = hIn   + (size_t)((tm << 4) + l) * K;
    const _Float16* wlCol  = WlT   + (size_t)((tn << 4) + l) * K;
    const _Float16* wrCol  = WrT   + (size_t)((tn << 4) + l) * K;

    float bv = bias[(tn << 4) + l];            // C layout: all 8 rows share N = l
    v8f acc = {bv, bv, bv, bv, bv, bv, bv, bv};

#pragma unroll
    for (int kb = 0; kb < K; kb += 32) {
        v16h aA = ld_a(aggRow, kb, g);
        v16h bL = ld_b(wlCol, kb, g);
        acc = __builtin_amdgcn_wmma_f32_16x16x32_f16(false, aA, false, bL,
                                                     (short)0, acc, false, false);
        v16h aX = ld_a(xRow, kb, g);
        v16h bR = ld_b(wrCol, kb, g);
        acc = __builtin_amdgcn_wmma_f32_16x16x32_f16(false, aX, false, bR,
                                                     (short)0, acc, false, false);
    }

    int nCol  = (tn << 4) + l;
    int mBase = (tm << 4) + (g << 3);          // lane group g -> rows M = 8g..8g+7
    if (fOut) {
#pragma unroll
        for (int r = 0; r < 8; ++r)
            fOut[(size_t)(mBase + r) * dOut + nCol] = acc[r];
    } else {
#pragma unroll
        for (int r = 0; r < 8; ++r) {
            float v = acc[r];
            v = v > 0.0f ? v : 0.1f * v;       // LeakyReLU(0.1)
            hOut[(size_t)(mBase + r) * dOut + nCol] = (_Float16)v;
        }
    }
}

// ---------------------------------------------------------------------------

extern "C" void kernel_launch(void* const* d_in, const int* in_sizes, int n_in,
                              void* d_out, int out_size, void* d_ws, size_t ws_size,
                              hipStream_t stream) {
    const float* x       = (const float*)d_in[0];
    const int*   eidx    = (const int*)d_in[1];
    const float* Wl_st   = (const float*)d_in[2];   // [3][128][128]
    const float* Wr_st   = (const float*)d_in[3];   // [3][128][128]
    const float* b_st    = (const float*)d_in[4];   // [3][128]
    const float* Wl_out  = (const float*)d_in[5];   // [128][64]
    const float* Wr_out  = (const float*)d_in[6];   // [128][64]
    const float* b_out   = (const float*)d_in[7];   // [64]
    float* out = (float*)d_out;

    const int* src = eidx;
    const int* dst = eidx + N_EDGES;

    // ---- carve workspace (assume d_ws 256B aligned) ----
    char* w = (char*)d_ws;
    size_t off = 0;
    auto alloc = [&](size_t bytes) -> void* {
        off = (off + 255) & ~(size_t)255;
        void* p = w + off;
        off += bytes;
        return p;
    };
    _Float16* h16A  = (_Float16*)alloc((size_t)N_NODES * DIM * 2);
    _Float16* h16B  = (_Float16*)alloc((size_t)N_NODES * DIM * 2);
    float*    agg32 = (float*)   alloc((size_t)N_NODES * DIM * 4);
    _Float16* agg16 = (_Float16*)alloc((size_t)N_NODES * DIM * 2);
    int*      cnt   = (int*)     alloc((size_t)N_NODES * 4);
    _Float16* WlT   = (_Float16*)alloc((size_t)3 * DIM * DIM * 2);  // [layer][n][k]
    _Float16* WrT   = (_Float16*)alloc((size_t)3 * DIM * DIM * 2);
    _Float16* WloT  = (_Float16*)alloc((size_t)DIM_OUT * DIM * 2);  // [n][k]
    _Float16* WroT  = (_Float16*)alloc((size_t)DIM_OUT * DIM * 2);
    (void)ws_size; (void)n_in; (void)in_sizes; (void)out_size;

    const int TPB = 256;

    // ---- weights: f32 row-major -> f16 column-major ----
    {
        int nW = DIM * DIM, blocks = (nW + TPB - 1) / TPB;
        for (int i = 0; i < 3; ++i) {
            k_transpose_w<<<blocks, TPB, 0, stream>>>(Wl_st + (size_t)i * nW, WlT + (size_t)i * nW, DIM, DIM);
            k_transpose_w<<<blocks, TPB, 0, stream>>>(Wr_st + (size_t)i * nW, WrT + (size_t)i * nW, DIM, DIM);
        }
        int nWo = DIM * DIM_OUT, blocksO = (nWo + TPB - 1) / TPB;
        k_transpose_w<<<blocksO, TPB, 0, stream>>>(Wl_out, WloT, DIM, DIM_OUT);
        k_transpose_w<<<blocksO, TPB, 0, stream>>>(Wr_out, WroT, DIM, DIM_OUT);
    }

    // ---- x -> f16 ----
    {
        int n = N_NODES * DIM;
        k_to_f16<<<(n + TPB - 1) / TPB, TPB, 0, stream>>>(x, h16A, n);
    }

    // ---- degree counts (dst-only, reused by all 4 layers) ----
    hipMemsetAsync(cnt, 0, (size_t)N_NODES * 4, stream);
    k_count<<<(N_EDGES + TPB - 1) / TPB, TPB, 0, stream>>>(dst, cnt, N_EDGES);

    const int nFeat      = N_NODES * DIM;
    const int featBlocks = (nFeat + TPB - 1) / TPB;
    const int scatBlocks = ((N_EDGES * 32) + TPB - 1) / TPB;

    _Float16* hCur  = h16A;
    _Float16* hNext = h16B;

    // ---- 3 hidden SAGE layers ----
    for (int layer = 0; layer < 3; ++layer) {
        hipMemsetAsync(agg32, 0, (size_t)nFeat * 4, stream);
        k_scatter<<<scatBlocks, TPB, 0, stream>>>(hCur, src, dst, agg32, N_EDGES);
        k_finalize<<<featBlocks, TPB, 0, stream>>>(agg32, cnt, agg16, nFeat);

        int totalTiles = (N_NODES / 16) * (DIM / 16);        // 50000
        int gemmBlocks = (totalTiles + 7) / 8;               // 8 waves / block
        k_sage_gemm<<<gemmBlocks, TPB, 0, stream>>>(
            hCur, agg16,
            WlT + (size_t)layer * DIM * DIM,
            WrT + (size_t)layer * DIM * DIM,
            b_st + (size_t)layer * DIM,
            hNext, nullptr, DIM, totalTiles);

        _Float16* t = hCur; hCur = hNext; hNext = t;
    }

    // ---- final SAGE layer (128 -> 64, no activation, f32 out) ----
    {
        hipMemsetAsync(agg32, 0, (size_t)nFeat * 4, stream);
        k_scatter<<<scatBlocks, TPB, 0, stream>>>(hCur, src, dst, agg32, N_EDGES);
        k_finalize<<<featBlocks, TPB, 0, stream>>>(agg32, cnt, agg16, nFeat);

        int totalTiles = (N_NODES / 16) * (DIM_OUT / 16);    // 25000
        int gemmBlocks = (totalTiles + 7) / 8;
        k_sage_gemm<<<gemmBlocks, TPB, 0, stream>>>(
            hCur, agg16, WloT, WroT, b_out,
            nullptr, out, DIM_OUT, totalTiles);
    }
}
